// RiskPredictionGNN_59846074303064
// MI455X (gfx1250) — matrix operand order repaired and checked by reading, not compiled, not run on hardware
//
#include <hip/hip_runtime.h>
#include <hip/hip_bf16.h>

// ---------------------------------------------------------------------------
// RiskPredictionGNN for MI455X (gfx1250): 2x GCNConv + BN + ReLU + MLP head.
// GEMMs: one wave per 16-row tile, all column tiles accumulated in registers
// via independent v_wmma_f32_16x16x32_bf16 issues (A fragment loaded once per
// k-step, reused across all column tiles -> 8x less A traffic, no WMMA->WMMA
// accumulator RAW chains). Edge aggregation uses f32 atomics (L2-resident).
// ---------------------------------------------------------------------------

typedef __attribute__((ext_vector_type(16))) __bf16 v16bf;
typedef __attribute__((ext_vector_type(8)))  float  v8f;

#define NN   100000
#define EE   1600000
#define DIN  64
#define DH   128
#define BNEPS 1e-5f

__device__ __forceinline__ unsigned short f2bf(float f) {
    union { float f; unsigned int u; } x; x.f = f;
    unsigned int u = x.u;
    u += 0x7FFFu + ((u >> 16) & 1u);   // round-to-nearest-even
    return (unsigned short)(u >> 16);
}

// ---------------- utility fills ----------------
__global__ void k_fill(float* __restrict__ p, float v, size_t n) {
    size_t i = (size_t)blockIdx.x * blockDim.x + threadIdx.x;
    size_t stride = (size_t)gridDim.x * blockDim.x;
    for (; i < n; i += stride) p[i] = v;
}

// ---------------- degree / dinv ----------------
__global__ void k_degree(const long long* __restrict__ dst, float* __restrict__ deg, int E) {
    int e = blockIdx.x * blockDim.x + threadIdx.x;
    if (e < E) atomicAdd(&deg[(int)dst[e]], 1.0f);
}

__global__ void k_dinv(const float* __restrict__ deg, float* __restrict__ dinv, int n) {
    int i = blockIdx.x * blockDim.x + threadIdx.x;
    if (i < n) dinv[i] = rsqrtf(deg[i]);
}

// ---------------- f32 -> bf16 pack (layer-1 input) ----------------
__global__ void k_pack_bf16(const float* __restrict__ src, unsigned short* __restrict__ dst, size_t n) {
    size_t i = (size_t)blockIdx.x * blockDim.x + threadIdx.x;
    size_t stride = (size_t)gridDim.x * blockDim.x;
    for (; i < n; i += stride) dst[i] = f2bf(src[i]);
}

// ---------------- pack weight W[K,Dout] into WMMA B-fragment order ----------------
// Layout: [tn][ks][lane][16] bf16, 32B per lane entry.
// B (32x16 per k-step): lane l -> column (l&15); K = ks*32 + (j>>3)*16 + (l>>4)*8 + (j&7).
__global__ void k_pack_bfrag(const float* __restrict__ W, unsigned short* __restrict__ Bp,
                             int K, int Dout) {
    int idx = blockIdx.x * blockDim.x + threadIdx.x;
    int total = K * Dout;
    if (idx >= total) return;
    int ksteps = K >> 5;
    int j  = idx & 15;
    int l  = (idx >> 4) & 31;
    int rest = idx >> 9;             // = tn*ksteps + ks
    int ks = rest % ksteps;
    int tn = rest / ksteps;
    int n  = tn * 16 + (l & 15);
    int k  = ks * 32 + ((j >> 3) << 4) + ((l >> 4) << 3) + (j & 7);
    Bp[idx] = f2bf(W[k * Dout + n]);
}

// ---------------- WMMA GEMM: C[M,Dout] = A[M,K](bf16) @ Bp(packed bf16) ----------------
// One wave per 16-row tile; DOUTT independent accumulators; A frag shared
// across all column tiles per k-step.
union AFrag { v16bf v; unsigned short s[16]; };

template<int KSTEPS, int DOUTT>
__global__ void k_gemm_wmma(const unsigned short* __restrict__ A,
                            const unsigned short* __restrict__ Bp,
                            float* __restrict__ C,
                            const float* __restrict__ bias,   // nullptr -> none
                            int M, int relu) {
    constexpr int K    = KSTEPS * 32;
    constexpr int Dout = DOUTT * 16;
    const int lane = threadIdx.x & 31;
    const int tm   = (blockIdx.x * blockDim.x + threadIdx.x) >> 5;  // wave = row tile
    if (tm >= (M >> 4)) return;

    // A-fragment addressing per ISA: lane<16 -> kb=0 (K 0..7 | 16..23),
    //                                lane>=16 -> kb=8 (K 8..15 | 24..31)
    const int row = tm * 16 + (lane & 15);
    const int kb  = (lane >> 4) << 3;
    const unsigned short* __restrict__ arow = A + (size_t)row * K;
    const unsigned short* __restrict__ blane = Bp + (size_t)lane * 16;

    v8f acc[DOUTT];
#pragma unroll
    for (int t = 0; t < DOUTT; ++t) acc[t] = (v8f){};

#pragma unroll
    for (int ks = 0; ks < KSTEPS; ++ks) {
        AFrag fa;
        const unsigned short* ap = arow + ks * 32 + kb;
#pragma unroll
        for (int i = 0; i < 8; ++i) { fa.s[i] = ap[i]; fa.s[8 + i] = ap[16 + i]; }
#pragma unroll
        for (int tn = 0; tn < DOUTT; ++tn) {
            v16bf fb = *(const v16bf*)(blane + ((size_t)tn * KSTEPS + ks) * 32 * 16);
            acc[tn] = __builtin_amdgcn_wmma_f32_16x16x32_bf16(
                          /*neg_a=*/false, fa.v, /*neg_b=*/false, fb,
                          /*c_mod=*/(short)0, acc[tn],
                          /*reuse_a=*/false, /*reuse_b=*/false);
        }
    }

    // C/D layout: VGPR r -> row r (lanes 0..15) or 8+r (lanes 16..31); col = lane&15
    const int orow  = tm * 16 + (lane >> 4) * 8;
    const int ocol0 = lane & 15;
#pragma unroll
    for (int tn = 0; tn < DOUTT; ++tn) {
        const int ocol = tn * 16 + ocol0;
        float badd = bias ? bias[ocol] : 0.0f;
#pragma unroll
        for (int r = 0; r < 8; ++r) {
            float v = acc[tn][r] + badd;
            if (relu) v = v > 0.0f ? v : 0.0f;
            C[(size_t)(orow + r) * Dout + ocol] = v;
        }
    }
}

// ---------------- edge scatter: agg[dst] += h[src] * dinv[src]*dinv[dst] ----------------
// One wave per edge; each lane covers D/32 channels with f32 atomics (L2-resident).
__global__ void k_scatter(const long long* __restrict__ src,
                          const long long* __restrict__ dst,
                          const float* __restrict__ h,
                          const float* __restrict__ dinv,
                          float* __restrict__ agg,
                          int E, int D) {
    size_t gtid = (size_t)blockIdx.x * blockDim.x + threadIdx.x;
    int e    = (int)(gtid >> 5);
    int lane = (int)(gtid & 31);
    if (e >= E) return;
    int s = (int)src[e], d = (int)dst[e];
    float coef = dinv[s] * dinv[d];
    const float* __restrict__ hs = h + (size_t)s * D;
    float* __restrict__ ad = agg + (size_t)d * D;
    for (int j = lane; j < D; j += 32)
        atomicAdd(ad + j, hs[j] * coef);
}

// ---------------- self-loop + bias, fused with BN stat accumulation ----------------
// blockDim.x == D (128): thread = channel; per-block partial sums -> 2 atomics/thread.
__global__ void k_finish_stats(float* __restrict__ agg,          // in: agg, out: t
                               const float* __restrict__ h,      // gemm output
                               const float* __restrict__ dinv,
                               const float* __restrict__ bias,
                               float* __restrict__ sum, float* __restrict__ sumsq,
                               int n, int D) {
    int c = threadIdx.x;
    float b = bias[c];
    float s = 0.0f, s2 = 0.0f;
    for (int row = blockIdx.x; row < n; row += gridDim.x) {
        size_t i = (size_t)row * D + c;
        float di = dinv[row];
        float v = agg[i] + h[i] * di * di + b;
        agg[i] = v;
        s += v; s2 += v * v;
    }
    atomicAdd(&sum[c], s);
    atomicAdd(&sumsq[c], s2);
}

__global__ void k_bn_params(const float* __restrict__ sum, const float* __restrict__ sumsq,
                            const float* __restrict__ g, const float* __restrict__ be,
                            float* __restrict__ scale, float* __restrict__ shift,
                            int n, float eps) {
    int c = threadIdx.x;
    float m   = sum[c] / (float)n;
    float var = sumsq[c] / (float)n - m * m;
    float sc  = g[c] * rsqrtf(var + eps);
    scale[c] = sc;
    shift[c] = be[c] - m * sc;
}

// ---------------- fused BN + ReLU + (f32 store, bf16 pack for next GEMM) ----------------
__global__ void k_bn_apply(const float* __restrict__ t,
                           const float* __restrict__ scale,
                           const float* __restrict__ shift,
                           float* __restrict__ hout,
                           unsigned short* __restrict__ hbf,
                           size_t total, int D) {
    size_t i = (size_t)blockIdx.x * blockDim.x + threadIdx.x;
    size_t stride = (size_t)gridDim.x * blockDim.x;
    for (; i < total; i += stride) {
        int c = (int)(i % (size_t)D);
        float v = t[i] * scale[c] + shift[c];
        v = v > 0.0f ? v : 0.0f;
        hout[i] = v;
        hbf[i]  = f2bf(v);
    }
}

// ---------------- head GEMV: risk = sigmoid(hidden[N,64] @ Wh2[64] + bh2) ----------------
__global__ void k_head(const float* __restrict__ hid,
                       const float* __restrict__ w, const float* __restrict__ b,
                       float* __restrict__ out, int n) {
    size_t gtid = (size_t)blockIdx.x * blockDim.x + threadIdx.x;
    int node = (int)(gtid >> 5);
    int lane = (int)(gtid & 31);
    if (node >= n) return;
    const float* __restrict__ hr = hid + (size_t)node * 64;
    float acc = hr[lane] * w[lane] + hr[lane + 32] * w[lane + 32];
#pragma unroll
    for (int off = 16; off > 0; off >>= 1) acc += __shfl_xor(acc, off, 32);
    if (lane == 0) out[node] = 1.0f / (1.0f + expf(-(acc + b[0])));
}

// ---------------------------------------------------------------------------
extern "C" void kernel_launch(void* const* d_in, const int* in_sizes, int n_in,
                              void* d_out, int out_size, void* d_ws, size_t ws_size,
                              hipStream_t stream) {
    const float*     x   = (const float*)d_in[0];
    const long long* ei  = (const long long*)d_in[1];   // [2,E] int64
    const float*     W1  = (const float*)d_in[2];
    const float*     b1  = (const float*)d_in[3];
    const float*     g1  = (const float*)d_in[4];
    const float*     be1 = (const float*)d_in[5];
    const float*     W2  = (const float*)d_in[6];
    const float*     b2  = (const float*)d_in[7];
    const float*     g2  = (const float*)d_in[8];
    const float*     be2 = (const float*)d_in[9];
    const float*     Wh1 = (const float*)d_in[10];
    const float*     bh1 = (const float*)d_in[11];
    const float*     Wh2 = (const float*)d_in[12];
    const float*     bh2 = (const float*)d_in[13];
    float* out = (float*)d_out;

    const long long* srcI = ei;        // edge_index[0]
    const long long* dstI = ei + EE;   // edge_index[1]

    // ---- workspace layout (d_ws is >=256B aligned) ----
    float* deg   = (float*)d_ws;                        // N
    float* dinv  = deg + NN;                            // N
    float* sum   = dinv + NN;                           // 128
    float* sumsq = sum + DH;                            // 128
    float* scale = sumsq + DH;                          // 128
    float* shift = scale + DH;                          // 128
    unsigned short* Wp  = (unsigned short*)(shift + DH);    // <=128*128 bf16 (32B-aligned)
    unsigned short* Abf = Wp + DH * DH;                     // N*128 bf16
    float* H    = (float*)(Abf + (size_t)NN * DH);          // N*128 f32 (gemm out / activations)
    float* AGG  = H + (size_t)NN * DH;                      // N*128 f32 (agg / t / head hidden)

    const int TB = 256;
    auto blocksFor = [](size_t n, int tb) { return (unsigned)((n + tb - 1) / tb); };
    const unsigned gemmBlk = blocksFor((size_t)(NN / 16) * 32, TB);  // one wave per row tile

    // ---- degree (with self-loop) and dinv ----
    k_fill<<<1024, TB, 0, stream>>>(deg, 1.0f, (size_t)NN);
    k_degree<<<blocksFor(EE, TB), TB, 0, stream>>>(dstI, deg, EE);
    k_dinv<<<blocksFor(NN, TB), TB, 0, stream>>>(deg, dinv, NN);

    // ---- layer 1: h = x @ W1 ----
    k_pack_bf16<<<4096, TB, 0, stream>>>(x, Abf, (size_t)NN * DIN);
    k_pack_bfrag<<<blocksFor((size_t)DIN * DH, TB), TB, 0, stream>>>(W1, Wp, DIN, DH);
    k_gemm_wmma<2, 8><<<gemmBlk, TB, 0, stream>>>(Abf, Wp, H, nullptr, NN, 0);

    k_fill<<<4096, TB, 0, stream>>>(AGG, 0.0f, (size_t)NN * DH);
    k_scatter<<<blocksFor((size_t)EE * 32, TB), TB, 0, stream>>>(srcI, dstI, H, dinv, AGG, EE, DH);

    k_fill<<<1, TB, 0, stream>>>(sum, 0.0f, (size_t)2 * DH);
    k_finish_stats<<<512, DH, 0, stream>>>(AGG, H, dinv, b1, sum, sumsq, NN, DH);
    k_bn_params<<<1, DH, 0, stream>>>(sum, sumsq, g1, be1, scale, shift, NN, BNEPS);
    k_bn_apply<<<4096, TB, 0, stream>>>(AGG, scale, shift, H, Abf, (size_t)NN * DH, DH);

    // ---- layer 2: h = relu_bn(agg(h @ W2)) ----
    k_pack_bfrag<<<blocksFor((size_t)DH * DH, TB), TB, 0, stream>>>(W2, Wp, DH, DH);
    k_gemm_wmma<4, 8><<<gemmBlk, TB, 0, stream>>>(Abf, Wp, H, nullptr, NN, 0);

    k_fill<<<4096, TB, 0, stream>>>(AGG, 0.0f, (size_t)NN * DH);
    k_scatter<<<blocksFor((size_t)EE * 32, TB), TB, 0, stream>>>(srcI, dstI, H, dinv, AGG, EE, DH);

    k_fill<<<1, TB, 0, stream>>>(sum, 0.0f, (size_t)2 * DH);
    k_finish_stats<<<512, DH, 0, stream>>>(AGG, H, dinv, b2, sum, sumsq, NN, DH);
    k_bn_params<<<1, DH, 0, stream>>>(sum, sumsq, g2, be2, scale, shift, NN, BNEPS);
    k_bn_apply<<<4096, TB, 0, stream>>>(AGG, scale, shift, H, Abf, (size_t)NN * DH, DH);

    // ---- head: hidden = relu(h @ Wh1 + bh1); risk = sigmoid(hidden @ Wh2 + bh2) ----
    k_pack_bfrag<<<blocksFor((size_t)DH * (DH / 2), TB), TB, 0, stream>>>(Wh1, Wp, DH, DH / 2);
    k_gemm_wmma<4, 4><<<gemmBlk, TB, 0, stream>>>(Abf, Wp, AGG, bh1, NN, 1);
    k_head<<<blocksFor((size_t)NN * 32, TB), TB, 0, stream>>>(AGG, Wh2, bh2, out, NN);
}
